// EEGformer_62947040690622
// MI455X (gfx1250) — matrix operand order repaired and checked by reading, not compiled, not run on hardware
//
#include <hip/hip_runtime.h>
#include <math.h>

// ---------------- types ----------------
typedef __attribute__((ext_vector_type(16))) __bf16 v16bf;
typedef __attribute__((ext_vector_type(8)))  __bf16 bfx8;
typedef __attribute__((ext_vector_type(8)))  float  v8f;

#define NA   120
#define NB   2048
#define NBP1 2049
#define MDIM 128
#define NTOK (NA * NBP1)          // 245880
#define FF   512

__device__ __forceinline__ v8f zero8() {
  v8f z = {0.f,0.f,0.f,0.f,0.f,0.f,0.f,0.f};
  return z;
}

__device__ __forceinline__ v8f wmma_bf16(v16bf a, v16bf b, v8f c) {
  // D = A(16x32 bf16) * B(32x16 bf16) + C(16x16 f32)
  return __builtin_amdgcn_wmma_f32_16x16x32_bf16(false, a, false, b, (short)0, c, false, false);
}

// ---- CDNA5 async global->LDS copy (ASYNCcnt) -------------------------------
// LDS destination = low 32 bits of the generic pointer (flat LDS aperture keeps
// addr[31:0] as the LDS byte offset). 16B per lane per issue.
__device__ __forceinline__ void async_copy16(const __bf16* g, __bf16* l) {
  unsigned lo = (unsigned)(uintptr_t)l;
  asm volatile("global_load_async_to_lds_b128 %0, %1, off"
               :: "v"(lo), "v"(g) : "memory");
}
__device__ __forceinline__ void wait_async0() {
  asm volatile("s_wait_asynccnt 0x0" ::: "memory");
}

// A-fragment (16x32 bf16) from LDS, row stride strideE (bf16 elems, mult of 8).
// ISA layout: lanes 0-15 row=lane, K groups {0..7,16..23}; lanes 16-31 row=lane-16, K {8..15,24..31}.
__device__ __forceinline__ v16bf load_a_lds(const __bf16* Z, int strideE, int rowBase, int kt) {
  const int lane = threadIdx.x & 31;
  const int row  = rowBase + (lane & 15);
  const int kb   = (lane >> 4) << 3;               // 0 or 8
  const __bf16* p = Z + row * strideE + kt * 32 + kb;
  union { v16bf v; bfx8 h[2]; } u;
  u.h[0] = *(const bfx8*)(p);
  u.h[1] = *(const bfx8*)(p + 16);
  return u.v;
}

// A-fragment from global fp32 matrix (row stride 128), converted to bf16 in regs.
__device__ __forceinline__ v16bf load_a_f32(const float* Mtx, int rowBase, int rowMax, int kt) {
  const int lane = threadIdx.x & 31;
  int row = rowBase + (lane & 15);
  if (row > rowMax) row = rowMax;                  // clamp stays within this block's rows
  const int kb = (lane >> 4) << 3;
  const float* p = Mtx + (size_t)row * MDIM + kt * 32 + kb;
  float4 f0 = *(const float4*)(p);
  float4 f1 = *(const float4*)(p + 4);
  float4 f2 = *(const float4*)(p + 16);
  float4 f3 = *(const float4*)(p + 20);
  v16bf r;
  r[0]=(__bf16)f0.x; r[1]=(__bf16)f0.y; r[2]=(__bf16)f0.z; r[3]=(__bf16)f0.w;
  r[4]=(__bf16)f1.x; r[5]=(__bf16)f1.y; r[6]=(__bf16)f1.z; r[7]=(__bf16)f1.w;
  r[8]=(__bf16)f2.x; r[9]=(__bf16)f2.y; r[10]=(__bf16)f2.z; r[11]=(__bf16)f2.w;
  r[12]=(__bf16)f3.x; r[13]=(__bf16)f3.y; r[14]=(__bf16)f3.z; r[15]=(__bf16)f3.w;
  return r;
}

// B-fragment (32x16 bf16), row-major [N][K] source (GEMM computes act @ W^T).
// ISA layout: lanes 0-15 col=lane, K groups {0..7,16..23}; lanes 16-31 col=lane-16, K {8..15,24..31}.
__device__ __forceinline__ v16bf load_b_bf16(const __bf16* W, int K, int nt, int kt) {
  const int lane = threadIdx.x & 31;
  const int n  = nt * 16 + (lane & 15);
  const int kb = (lane >> 4) << 3;                 // 0 or 8
  const __bf16* p = W + (size_t)n * K + kt * 32 + kb;
  union { v16bf v; bfx8 h[2]; } u;
  u.h[0] = *(const bfx8*)(p);
  u.h[1] = *(const bfx8*)(p + 16);
  return u.v;
}

// Same, but from an LDS panel with padded row stride (bank spread).
__device__ __forceinline__ v16bf load_b_lds(const __bf16* Wb, int strideE, int nt, int kt) {
  const int lane = threadIdx.x & 31;
  const int n  = nt * 16 + (lane & 15);
  const int kb = (lane >> 4) << 3;
  const __bf16* p = Wb + n * strideE + kt * 32 + kb;
  union { v16bf v; bfx8 h[2]; } u;
  u.h[0] = *(const bfx8*)(p);
  u.h[1] = *(const bfx8*)(p + 16);
  return u.v;
}

// ---------------- weight fp32 -> bf16 staging ----------------
__global__ __launch_bounds__(256) void k_cvt(const float* __restrict__ src,
                                             __bf16* __restrict__ dst, int n) {
  int i = blockIdx.x * 256 + threadIdx.x;
  if (i < n) dst[i] = (__bf16)src[i];
}

// ---------------- s[a,0,:] = cls + bias ----------------
__global__ __launch_bounds__(128) void k_cls(const float* __restrict__ cls,
                                             const float* __restrict__ bias,
                                             float* __restrict__ s) {
  int a = blockIdx.x, c = threadIdx.x;
  size_t g = (size_t)a * NBP1 * MDIM + c;
  s[g] = cls[(size_t)a * MDIM + c] + bias[g];
}

// ---------------- s[a,1+j,:] = x @ W^T + bias ----------------
__global__ __launch_bounds__(256) void k_embed(const float* __restrict__ x,
                                               const __bf16* __restrict__ wgt,
                                               const float* __restrict__ bias,
                                               float* __restrict__ s) {
  const int a  = blockIdx.x >> 4;                  // 16 tiles of 128 tokens per sequence
  const int jb = blockIdx.x & 15;
  const int w = threadIdx.x >> 5, lane = threadIdx.x & 31;
  const int rowBase = w * 16;
  const int hi = lane >> 4, col0 = lane & 15;
  const float* xt = x + ((size_t)a * NB + (size_t)jb * 128) * MDIM;
  v16bf af[4];
  #pragma unroll
  for (int kt = 0; kt < 4; ++kt) af[kt] = load_a_f32(xt, rowBase, 127, kt);
  #pragma unroll
  for (int nt = 0; nt < 8; ++nt) {
    v8f acc = zero8();
    #pragma unroll
    for (int kt = 0; kt < 4; ++kt)
      acc = wmma_bf16(af[kt], load_b_bf16(wgt, MDIM, nt, kt), acc);
    #pragma unroll
    for (int r = 0; r < 8; ++r) {
      int row = rowBase + r + hi * 8;
      size_t g = ((size_t)a * NBP1 + 1 + jb * 128 + row) * MDIM + nt * 16 + col0;
      s[g] = acc[r] + bias[g];
    }
  }
}

// ---------------- LN1 + QKV + per-head q.k*v -> imv (pre-scan) ----------------
// QKV weight panel (384x128 bf16) is staged into LDS with async copies that
// overlap the LN1 phase; all B-fragments then come from LDS.
__global__ __launch_bounds__(256) void k_qkv(const float* __restrict__ s,
                                             const __bf16* __restrict__ wqkv,
                                             const float* __restrict__ g1,
                                             const float* __restrict__ b1,
                                             float* __restrict__ imv, int Ntok) {
  __shared__ __bf16 Z[128 * 136];                  // 34.8 KB LN1 output
  __shared__ __bf16 W[384 * 136];                  // 104.4 KB qkv panel (padded rows)
  const int tok0 = blockIdx.x * 128;
  const int valid = (Ntok - tok0 < 128) ? (Ntok - tok0) : 128;
  const int w = threadIdx.x >> 5, lane = threadIdx.x & 31;

  // kick off async weight staging: 384 rows x 16 chunks of 16B
  for (int i = threadIdx.x; i < 384 * 16; i += 256) {
    int r = i >> 4, c = i & 15;
    async_copy16(wqkv + (size_t)r * MDIM + c * 8, &W[r * 136 + c * 8]);
  }

  // LN1: one wave per token, 4 channels per lane, wave32 shuffle reduction
  float4 gv = *(const float4*)(g1 + lane * 4);
  float4 bv = *(const float4*)(b1 + lane * 4);
  for (int it = 0; it < 16; ++it) {
    int t = it * 8 + w;
    if (t < valid) {
      const float4 v = *(const float4*)(s + (size_t)(tok0 + t) * MDIM + lane * 4);
      float sm = v.x + v.y + v.z + v.w;
      float sq = v.x*v.x + v.y*v.y + v.z*v.z + v.w*v.w;
      #pragma unroll
      for (int o = 1; o < 32; o <<= 1) { sm += __shfl_xor(sm, o, 32); sq += __shfl_xor(sq, o, 32); }
      float mu = sm * (1.f / 128.f);
      float rs = rsqrtf(sq * (1.f / 128.f) - mu * mu + 1e-5f);
      __bf16* zp = Z + t * 136 + lane * 4;
      zp[0] = (__bf16)((v.x - mu) * rs * gv.x + bv.x);
      zp[1] = (__bf16)((v.y - mu) * rs * gv.y + bv.y);
      zp[2] = (__bf16)((v.z - mu) * rs * gv.z + bv.z);
      zp[3] = (__bf16)((v.w - mu) * rs * gv.w + bv.w);
    }
  }
  wait_async0();
  __syncthreads();

  const int rowBase = w * 16;
  const int hi = lane >> 4, col0 = lane & 15;
  const float inv_sqrt_dh = 0.25f;                 // 1/sqrt(16)
  #pragma unroll 1
  for (int h = 0; h < 8; ++h) {                    // each 16-col N-tile == one head
    v8f aq = zero8(), ak = zero8(), av = zero8();
    #pragma unroll
    for (int kt = 0; kt < 4; ++kt) {
      v16bf afr = load_a_lds(Z, 136, rowBase, kt);
      aq = wmma_bf16(afr, load_b_lds(W,             136, h, kt), aq);
      ak = wmma_bf16(afr, load_b_lds(W + 128 * 136, 136, h, kt), ak);
      av = wmma_bf16(afr, load_b_lds(W + 256 * 136, 136, h, kt), av);
    }
    #pragma unroll
    for (int r = 0; r < 8; ++r) {
      float p = aq[r] * ak[r];                     // reduce q.k across the 16 head-dims (lanes)
      p += __shfl_xor(p, 1, 32); p += __shfl_xor(p, 2, 32);
      p += __shfl_xor(p, 4, 32); p += __shfl_xor(p, 8, 32);
      float val = p * inv_sqrt_dh * av[r];
      int row = rowBase + r + hi * 8;
      if (row < valid)
        imv[(size_t)(tok0 + row) * MDIM + h * 16 + col0] = val;
    }
  }
}

// ---------------- two-phase prefix sum over j in [0,2048) per sequence/channel ----------------
__global__ __launch_bounds__(128) void k_scan1(float* __restrict__ imv, float* __restrict__ chks) {
  int a = blockIdx.x >> 4, ch = blockIdx.x & 15, c = threadIdx.x;
  size_t base = ((size_t)a * NBP1 + (size_t)ch * 128) * MDIM + c;
  float run = 0.f;
  for (int j = 0; j < 128; ++j) { size_t g = base + (size_t)j * MDIM; run += imv[g]; imv[g] = run; }
  chks[(size_t)blockIdx.x * MDIM + c] = run;
}

__global__ __launch_bounds__(128) void k_scan2(float* __restrict__ imv, const float* __restrict__ chks) {
  int a = blockIdx.x >> 4, ch = blockIdx.x & 15, c = threadIdx.x;
  if (ch == 0) return;
  float off = 0.f;
  for (int q = 0; q < ch; ++q) off += chks[((size_t)(a * 16 + q)) * MDIM + c];
  size_t base = ((size_t)a * NBP1 + (size_t)ch * 128) * MDIM + c;
  for (int j = 0; j < 128; ++j) imv[base + (size_t)j * MDIM] += off;
}

// ---------------- Wo+residual -> LN2 -> fc1+GELU -> fc2+residual ----------------
__global__ __launch_bounds__(256) void k_mlp(float* __restrict__ s,
                                             const float* __restrict__ imv,
                                             const __bf16* __restrict__ wo,
                                             const __bf16* __restrict__ wf1,
                                             const __bf16* __restrict__ wf2,
                                             const float* __restrict__ g2,
                                             const float* __restrict__ b2,
                                             const float* __restrict__ f1b,
                                             const float* __restrict__ f2b, int Ntok) {
  __shared__ float  S[32 * 132];                   // post-Wo residual stream (fp32)
  __shared__ __bf16 Z[32 * 136];                   // LN2 output (bf16)
  __shared__ __bf16 H[32 * 520];                   // GELU(fc1) output (bf16)
  __shared__ __bf16 WO[128 * 136];                 // Wo panel (padded rows)
  const int tok0 = blockIdx.x * 32;
  const int w = threadIdx.x >> 5, lane = threadIdx.x & 31;
  const int rt = w & 1, wg = w >> 1;
  const int rowBase = rt * 16;
  const int hi = lane >> 4, col0 = lane & 15;
  const int rowMaxL = (Ntok - 1) - tok0;           // local clamp bound (within this block)

  // async-stage Wo panel (128 rows x 16 chunks of 16B), overlapped with A-frag loads
  for (int i = threadIdx.x; i < 128 * 16; i += 256) {
    int r = i >> 4, c = i & 15;
    async_copy16(wo + (size_t)r * MDIM + c * 8, &WO[r * 136 + c * 8]);
  }

  // Stage 1: s_new = cumsum_imv @ Wo^T + s_old  -> LDS S
  {
    v16bf af[4];
    #pragma unroll
    for (int kt = 0; kt < 4; ++kt)
      af[kt] = load_a_f32(imv + (size_t)tok0 * MDIM, rowBase, rowMaxL, kt);
    wait_async0();
    __syncthreads();
    #pragma unroll
    for (int q = 0; q < 2; ++q) {
      int nt = wg + q * 4;
      v8f acc = zero8();
      #pragma unroll
      for (int kt = 0; kt < 4; ++kt)
        acc = wmma_bf16(af[kt], load_b_lds(WO, 136, nt, kt), acc);
      #pragma unroll
      for (int r = 0; r < 8; ++r) {
        int row = rowBase + r + hi * 8;
        int grow = row; if (grow > rowMaxL) grow = rowMaxL;
        int col = nt * 16 + col0;
        S[row * 132 + col] = acc[r] + s[(size_t)(tok0 + grow) * MDIM + col];
      }
    }
  }
  __syncthreads();

  // Stage 2: LN2 -> Z (bf16)
  {
    float4 gv = *(const float4*)(g2 + lane * 4);
    float4 bv = *(const float4*)(b2 + lane * 4);
    #pragma unroll
    for (int it = 0; it < 4; ++it) {
      int t = w * 4 + it;
      const float4 v = *(const float4*)(S + t * 132 + lane * 4);
      float sm = v.x + v.y + v.z + v.w;
      float sq = v.x*v.x + v.y*v.y + v.z*v.z + v.w*v.w;
      #pragma unroll
      for (int o = 1; o < 32; o <<= 1) { sm += __shfl_xor(sm, o, 32); sq += __shfl_xor(sq, o, 32); }
      float mu = sm * (1.f / 128.f);
      float rs = rsqrtf(sq * (1.f / 128.f) - mu * mu + 1e-5f);
      __bf16* zp = Z + t * 136 + lane * 4;
      zp[0] = (__bf16)((v.x - mu) * rs * gv.x + bv.x);
      zp[1] = (__bf16)((v.y - mu) * rs * gv.y + bv.y);
      zp[2] = (__bf16)((v.z - mu) * rs * gv.z + bv.z);
      zp[3] = (__bf16)((v.w - mu) * rs * gv.w + bv.w);
    }
  }
  __syncthreads();

  // Stage 3: H = GELU(Z @ fc1^T + b1)  (N=512)
  {
    v16bf af[4];
    #pragma unroll
    for (int kt = 0; kt < 4; ++kt) af[kt] = load_a_lds(Z, 136, rowBase, kt);
    #pragma unroll 1
    for (int q = 0; q < 8; ++q) {
      int nt = wg * 8 + q;
      v8f acc = zero8();
      #pragma unroll
      for (int kt = 0; kt < 4; ++kt)
        acc = wmma_bf16(af[kt], load_b_bf16(wf1, MDIM, nt, kt), acc);
      #pragma unroll
      for (int r = 0; r < 8; ++r) {
        int col = nt * 16 + col0;
        float v = acc[r] + f1b[col];
        float gl = 0.5f * v * (1.f + erff(v * 0.70710678f));
        H[(rowBase + r + hi * 8) * 520 + col] = (__bf16)gl;
      }
    }
  }
  __syncthreads();

  // Stage 4: s = H @ fc2^T + b2 + s_new  (K=512)
  {
    #pragma unroll 1
    for (int q = 0; q < 2; ++q) {
      int nt = wg + q * 4;
      v8f acc = zero8();
      #pragma unroll
      for (int kt = 0; kt < 16; ++kt)
        acc = wmma_bf16(load_a_lds(H, 520, rowBase, kt), load_b_bf16(wf2, FF, nt, kt), acc);
      #pragma unroll
      for (int r = 0; r < 8; ++r) {
        int row = rowBase + r + hi * 8;
        int grow = tok0 + row;
        int col = nt * 16 + col0;
        if (grow < Ntok)
          s[(size_t)grow * MDIM + col] = acc[r] + f2b[col] + S[row * 132 + col];
      }
    }
  }
}

// ---------------- host ----------------
extern "C" void kernel_launch(void* const* d_in, const int* in_sizes, int n_in,
                              void* d_out, int out_size, void* d_ws, size_t ws_size,
                              hipStream_t stream) {
  const float* x      = (const float*)d_in[0];
  const float* weight = (const float*)d_in[1];
  const float* cls    = (const float*)d_in[2];
  const float* bias   = (const float*)d_in[3];
  const float* Wqkv   = (const float*)d_in[4];
  const float* Wo     = (const float*)d_in[5];
  const float* g1     = (const float*)d_in[6];
  const float* b1     = (const float*)d_in[7];
  const float* g2     = (const float*)d_in[8];
  const float* b2     = (const float*)d_in[9];
  const float* f1w    = (const float*)d_in[10];
  const float* f1b    = (const float*)d_in[11];
  const float* f2w    = (const float*)d_in[12];
  const float* f2b    = (const float*)d_in[13];
  float* s = (float*)d_out;

  // workspace: imv (NTOK*128 f32) | chunk sums (120*16*128 f32) | bf16 weights
  float*  imv  = (float*)d_ws;
  float*  chks = imv + (size_t)NTOK * MDIM;
  __bf16* wb   = (__bf16*)(chks + (size_t)NA * 16 * MDIM);
  __bf16* wb_wgt = wb;                              // 128x128
  __bf16* wb_qkv = wb_wgt + 16384;                  // 2 x (3*128)x128
  __bf16* wb_o   = wb_qkv + 2 * 49152;              // 2 x 128x128
  __bf16* wb_f1  = wb_o   + 2 * 16384;              // 512x128
  __bf16* wb_f2  = wb_f1  + 65536;                  // 128x512

  k_cvt<<<(16384 + 255) / 256, 256, 0, stream>>>(weight, wb_wgt, 16384);
  k_cvt<<<(98304 + 255) / 256, 256, 0, stream>>>(Wqkv,   wb_qkv, 98304);
  k_cvt<<<(32768 + 255) / 256, 256, 0, stream>>>(Wo,     wb_o,   32768);
  k_cvt<<<(65536 + 255) / 256, 256, 0, stream>>>(f1w,    wb_f1,  65536);
  k_cvt<<<(65536 + 255) / 256, 256, 0, stream>>>(f2w,    wb_f2,  65536);

  k_cls<<<NA, 128, 0, stream>>>(cls, bias, s);
  k_embed<<<NA * 16, 256, 0, stream>>>(x, wb_wgt, bias, s);

  for (int layer = 0; layer < 2; ++layer) {
    k_qkv<<<(NTOK + 127) / 128, 256, 0, stream>>>(s, wb_qkv + layer * 49152, g1, b1, imv, NTOK);
    k_scan1<<<NA * 16, 128, 0, stream>>>(imv, chks);
    k_scan2<<<NA * 16, 128, 0, stream>>>(imv, chks);
    k_mlp<<<(NTOK + 31) / 32, 256, 0, stream>>>(s, imv, wb_o + layer * 16384,
                                                wb_f1, wb_f2, g2, b2, f1b, f2b, NTOK);
  }
}
